// PCVRSymbiosis_42786464203261
// MI455X (gfx1250) — matrix operand order repaired and checked by reading, not compiled, not run on hardware
//
#include <hip/hip_runtime.h>

#define B_    4
#define S_LEN 1024
#define L_LEN 1024
#define D_    1024
#define H_    16
#define HD_   64
#define HID_  4096
#define M_    (B_*S_LEN)

typedef __bf16 bf;
typedef __attribute__((ext_vector_type(16))) __bf16 v16bf;
typedef __attribute__((ext_vector_type(8)))  __bf16 v8bf;
typedef __attribute__((ext_vector_type(8)))  float  v8f;

union bfu { v16bf v; v8bf h[2]; };

// ---- WMMA fragment helpers (CDNA5 16x16x32 bf16 layouts, wave32) ----
// A 16x32: lanes 0-15 row M=lane, halves = K{16*hs .. 16*hs+15} (contiguous)
__device__ inline v16bf load_a_frag(const bf* rowk0, int hs) {
  bfu u;
  u.h[0] = *(const v8bf*)(rowk0 + 16 * hs);
  u.h[1] = *(const v8bf*)(rowk0 + 16 * hs + 8);
  return u.v;
}
// B 32x16: lane col N=lane&15, halves = K{16*hs .. 16*hs+15} (contiguous)
__device__ inline v16bf load_b_frag(const bf* p) {
  bfu u;
  u.h[0] = *(const v8bf*)(p);
  u.h[1] = *(const v8bf*)(p + 8);
  return u.v;
}
__device__ inline v8f wmma_bf(v16bf a, v16bf b, v8f c) {
  return __builtin_amdgcn_wmma_f32_16x16x32_bf16(false, a, false, b, (short)0, c,
                                                 false, false);
}

// ---------------- small kernels ----------------
__global__ void f32_to_bf16_kernel(const float* __restrict__ in, bf* __restrict__ out,
                                   size_t n) {
  size_t i = (size_t)blockIdx.x * blockDim.x + threadIdx.x;
  if (i < n) out[i] = (bf)in[i];
}

// scale/shift = modulation @ film_w^T + film_b   (B x 2D)
__global__ void film_kernel(const float* __restrict__ mod, const float* __restrict__ fw,
                            const float* __restrict__ fb, float* __restrict__ ss) {
  int n = blockIdx.x, b = blockIdx.y, tid = threadIdx.x;
  __shared__ float red[256];
  float s = 0.f;
  for (int k = tid; k < D_; k += 256)
    s += mod[(size_t)b * D_ + k] * fw[(size_t)n * D_ + k];
  red[tid] = s; __syncthreads();
  for (int st = 128; st > 0; st >>= 1) {
    if (tid < st) red[tid] += red[tid + st];
    __syncthreads();
  }
  if (tid == 0) ss[(size_t)b * 2 * D_ + n] = red[0] + fb[n];
}

// validf = ~safe_key_padding_mask(mask); rowvalid = any(~mask) per batch
__global__ void mask_kernel(const unsigned char* __restrict__ mask,
                            float* __restrict__ validf, float* __restrict__ rowvalid,
                            int Lk) {
  int b = blockIdx.x, tid = threadIdx.x;
  __shared__ int red[256];
  int allpad = 1;
  for (int k = tid; k < Lk; k += 256)
    if (mask[(size_t)b * Lk + k] == 0) allpad = 0;
  red[tid] = allpad; __syncthreads();
  for (int st = 128; st > 0; st >>= 1) {
    if (tid < st) red[tid] &= red[tid + st];
    __syncthreads();
  }
  int ap = red[0];
  for (int k = tid; k < Lk; k += 256) {
    unsigned char m = mask[(size_t)b * Lk + k];
    validf[(size_t)b * Lk + k] = ap ? 1.f : (m ? 0.f : 1.f);
  }
  if (rowvalid && tid == 0) rowvalid[b] = ap ? 0.f : 1.f;
}

// rmsnorm (+ optional FiLM) -> bf16
__global__ void rmsnorm_kernel(const float* __restrict__ x, const float* __restrict__ w,
                               const float* __restrict__ ss, bf* __restrict__ out,
                               int rows_per_batch) {
  int row = blockIdx.x, tid = threadIdx.x;
  const float* xr = x + (size_t)row * D_;
  __shared__ float red[256];
  __shared__ float s_rinv;
  float s = 0.f;
  for (int k = tid; k < D_; k += 256) { float v = xr[k]; s += v * v; }
  red[tid] = s; __syncthreads();
  for (int st = 128; st > 0; st >>= 1) {
    if (tid < st) red[tid] += red[tid + st];
    __syncthreads();
  }
  if (tid == 0) s_rinv = rsqrtf(red[0] * (1.f / (float)D_) + 1e-6f);
  __syncthreads();
  float rinv = s_rinv;
  int b = row / rows_per_batch;
  for (int k = tid; k < D_; k += 256) {
    float v = xr[k] * rinv * w[k];
    if (ss) {
      float sc = ss[(size_t)b * 2 * D_ + k];
      float sh = ss[(size_t)b * 2 * D_ + D_ + k];
      v = v * (1.f + 0.1f * tanhf(sc)) + sh;
    }
    out[(size_t)row * D_ + k] = (bf)v;
  }
}

// qkv (f32, [M,3D]) -> rope(q),rope(k) [B,H,S,HD] bf16, v transposed [B,H,HD,S] bf16
__global__ void rope_split_kernel(const float* __restrict__ qkv, bf* __restrict__ qh,
                                  bf* __restrict__ kh, bf* __restrict__ vT,
                                  size_t total) {
  size_t t = (size_t)blockIdx.x * blockDim.x + threadIdx.x;
  if (t >= total) return;
  int pr = (int)(t & 31); size_t r = t >> 5;
  int h = (int)(r & 15); r >>= 4;
  int s = (int)(r & (S_LEN - 1)); int b = (int)(r >> 10);
  float freq = __expf(-(2.0f * pr) * (9.210340371976184f / (float)HD_)); // ln(1e4)
  float ang = (float)s * freq;
  float sn, cs; __sincosf(ang, &sn, &cs);
  size_t base = ((size_t)b * S_LEN + s) * (3 * D_) + (size_t)h * HD_ + 2 * pr;
  float qe = qkv[base], qo = qkv[base + 1];
  float ke = qkv[base + D_], ko = qkv[base + D_ + 1];
  float ve = qkv[base + 2 * D_], vo = qkv[base + 2 * D_ + 1];
  size_t hb = (((size_t)(b * H_ + h)) * S_LEN + s) * HD_ + 2 * pr;
  qh[hb] = (bf)(qe * cs - qo * sn);
  qh[hb + 1] = (bf)(qe * sn + qo * cs);
  kh[hb] = (bf)(ke * cs - ko * sn);
  kh[hb + 1] = (bf)(ke * sn + ko * cs);
  size_t vb0 = (((size_t)(b * H_ + h)) * HD_ + 2 * pr) * S_LEN + s;
  vT[vb0] = (bf)ve;
  vT[vb0 + S_LEN] = (bf)vo;
}

// cross: cq [M,D], ckv [M,2D] -> heads bf16 (no rope), v transposed
__global__ void cross_split_kernel(const float* __restrict__ cq,
                                   const float* __restrict__ ckv, bf* __restrict__ qh,
                                   bf* __restrict__ kh, bf* __restrict__ vT,
                                   size_t total) {
  size_t t = (size_t)blockIdx.x * blockDim.x + threadIdx.x;
  if (t >= total) return;
  int d = (int)(t & 63); size_t r = t >> 6;
  int h = (int)(r & 15); r >>= 4;
  int s = (int)(r & (L_LEN - 1)); int b = (int)(r >> 10);
  size_t hb = (((size_t)(b * H_ + h)) * L_LEN + s) * HD_ + d;
  qh[hb] = (bf)cq[((size_t)b * S_LEN + s) * D_ + h * HD_ + d];
  size_t kb = ((size_t)b * L_LEN + s) * (2 * D_) + h * HD_ + d;
  kh[hb] = (bf)ckv[kb];
  vT[(((size_t)(b * H_ + h)) * HD_ + d) * L_LEN + s] = (bf)ckv[kb + D_];
}

// ---------------- generic WMMA GEMM: C = A @ W^T (+bias) with fused epilogues ----
// A: [M,K] bf16 row-major (lda), W: [N,K] bf16 row-major (ldw)
// Block = 4 waves; each wave computes a 32(M) x 64(N) tile (2 A-frags x 4 B-frags,
// 8 accumulators) so every B fragment feeds two WMMAs.
// epi: 0 Cf=acc+b | 2 Cf=res+acc+b (Cb bf16 copy if non-null)
//      3 t=(acc+b)*rowscale[row/rpb]; Cf=t, Cb=t
//      4 Cf = res + sigmoid(aux+acc+b)*aux2
//      5 Cb = bf16( silu(acc+b) * aux )
__global__ __launch_bounds__(128) void gemm_bf16_kernel(
    const bf* __restrict__ A, int lda, const bf* __restrict__ W, int ldw,
    const float* __restrict__ bias, float* __restrict__ Cf, bf* __restrict__ Cb,
    int ldc, const float* __restrict__ res, const float* __restrict__ aux,
    const float* __restrict__ aux2, const float* __restrict__ rowscale,
    int rows_per_batch, int K, int epi) {
  const int lane = threadIdx.x & 31;
  const int wave = threadIdx.x >> 5;
  const int hs = lane >> 4;
  const int l16 = lane & 15;
  const int m_base = blockIdx.y * 128 + wave * 32;
  const int n_base = blockIdx.x * 64;

  v8f acc0[4] = {};
  v8f acc1[4] = {};
  const bf* Arow0 = A + (size_t)(m_base + l16) * lda;
  const bf* Arow1 = Arow0 + (size_t)16 * lda;
  const bf* W0 = W + (size_t)(n_base + l16) * ldw;
  const size_t wstep = (size_t)16 * ldw;

  for (int k0 = 0; k0 < K; k0 += 32) {
    __builtin_prefetch(Arow0 + k0 + 256, 0, 3);  // global_prefetch_b8 (near-temporal)
    __builtin_prefetch(Arow1 + k0 + 256, 0, 3);
    v16bf a0 = load_a_frag(Arow0 + k0, hs);
    v16bf a1 = load_a_frag(Arow1 + k0, hs);
    const bf* wp = W0 + k0 + 16 * hs;
#pragma unroll
    for (int j = 0; j < 4; j++) {
      v16bf b = load_b_frag(wp + j * wstep);
      acc0[j] = wmma_bf(a0, b, acc0[j]);
      acc1[j] = wmma_bf(a1, b, acc1[j]);
    }
  }

#pragma unroll
  for (int half = 0; half < 2; half++) {
    v8f* acc = half ? acc1 : acc0;
    const int mh = m_base + half * 16;
#pragma unroll
    for (int j = 0; j < 4; j++) {
      int col = n_base + j * 16 + l16;
      float bv = bias ? bias[col] : 0.f;
#pragma unroll
      for (int r = 0; r < 8; r++) {
        int row = mh + r + 8 * hs;
        size_t idx = (size_t)row * ldc + col;
        float v = acc[j][r] + bv;
        if (epi == 0) {
          Cf[idx] = v;
        } else if (epi == 2) {
          float t = res[idx] + v;
          Cf[idx] = t;
          if (Cb) Cb[idx] = (bf)t;
        } else if (epi == 3) {
          float t = v * rowscale[row / rows_per_batch];
          Cf[idx] = t;
          Cb[idx] = (bf)t;
        } else if (epi == 4) {
          float g = 1.f / (1.f + __expf(-(aux[idx] + v)));
          Cf[idx] = res[idx] + g * aux2[idx];
        } else if (epi == 5) {
          float s = v / (1.f + __expf(-v));
          Cb[idx] = (bf)(s * aux[idx]);
        }
      }
    }
  }
}

// ---------------- WMMA flash attention ----------------
// Q,K: [B,H,L,HD] bf16; Vt: [B,H,HD,L] bf16; validf: [B,L] (1=valid)
// Out: bf16 merged heads [B,S,D]
__global__ __launch_bounds__(128) void flash_attn_kernel(
    const bf* __restrict__ Q, const bf* __restrict__ Kb, const bf* __restrict__ Vt,
    const float* __restrict__ validf, bf* __restrict__ Ob, int Sq, int Lk) {
  __shared__ bf lds_p[4][16 * 32];
  const int lane = threadIdx.x & 31;
  const int wave = threadIdx.x >> 5;
  const int hs = lane >> 4;
  const int l16 = lane & 15;
  const int h = blockIdx.y, b = blockIdx.z;
  const int q0 = blockIdx.x * 64 + wave * 16;
  const size_t bh = (size_t)(b * H_ + h);

  const bf* Qrow = Q + (bh * Sq + q0 + l16) * HD_;
  v16bf a0 = load_a_frag(Qrow, hs);
  v16bf a1 = load_a_frag(Qrow + 32, hs);

  v8f o[4] = {};
  float mi[8], li[8];
#pragma unroll
  for (int r = 0; r < 8; r++) { mi[r] = -1e30f; li[r] = 0.f; }
  const float* vrow = validf + (size_t)b * Lk;

  for (int kt = 0; kt < Lk; kt += 32) {
    const bf* Kr0 = Kb + (bh * Lk + kt + l16) * HD_;
    const bf* Kr1 = Kr0 + (size_t)16 * HD_;
    v8f s0 = {}, s1 = {};
    s0 = wmma_bf(a0, load_b_frag(Kr0 + 16 * hs), s0);
    s0 = wmma_bf(a1, load_b_frag(Kr0 + 32 + 16 * hs), s0);
    s1 = wmma_bf(a0, load_b_frag(Kr1 + 16 * hs), s1);
    s1 = wmma_bf(a1, load_b_frag(Kr1 + 32 + 16 * hs), s1);

    float mad0 = vrow[kt + l16] > 0.5f ? 0.f : -3.0e38f;
    float mad1 = vrow[kt + 16 + l16] > 0.5f ? 0.f : -3.0e38f;

#pragma unroll
    for (int r = 0; r < 8; r++) {
      float x0 = s0[r] * 0.125f + mad0;  // 1/sqrt(64)
      float x1 = s1[r] * 0.125f + mad1;
      float mx = fmaxf(x0, x1);
#pragma unroll
      for (int off = 1; off < 16; off <<= 1) mx = fmaxf(mx, __shfl_xor(mx, off, 32));
      float mn = fmaxf(mi[r], mx);
      float al = __expf(mi[r] - mn);
      float p0 = __expf(x0 - mn);
      float p1 = __expf(x1 - mn);
      float rs = p0 + p1;
#pragma unroll
      for (int off = 1; off < 16; off <<= 1) rs += __shfl_xor(rs, off, 32);
      li[r] = li[r] * al + rs;
      mi[r] = mn;
      o[0][r] = o[0][r] * al;
      o[1][r] = o[1][r] * al;
      o[2][r] = o[2][r] * al;
      o[3][r] = o[3][r] * al;
      int prow = r + 8 * hs;
      lds_p[wave][prow * 32 + l16] = (bf)p0;
      lds_p[wave][prow * 32 + 16 + l16] = (bf)p1;
    }
    // C-layout -> A-layout through per-wave LDS tile (DS ops are in-order per wave)
    v16bf pf = load_a_frag(&lds_p[wave][l16 * 32], hs);
#pragma unroll
    for (int j = 0; j < 4; j++) {
      const bf* Vr = Vt + (bh * HD_ + j * 16 + l16) * Lk + kt + 16 * hs;
      o[j] = wmma_bf(pf, load_b_frag(Vr), o[j]);
    }
  }

#pragma unroll
  for (int j = 0; j < 4; j++)
#pragma unroll
    for (int r = 0; r < 8; r++) {
      int row = q0 + r + 8 * hs;
      int col = h * HD_ + j * 16 + l16;
      float v = o[j][r] / li[r];
      Ob[((size_t)b * Sq + row) * D_ + col] = (bf)v;
    }
}

// ---------------- host ----------------
extern "C" void kernel_launch(void* const* d_in, const int* in_sizes, int n_in,
                              void* d_out, int out_size, void* d_ws, size_t ws_size,
                              hipStream_t stream) {
  (void)in_sizes; (void)n_in; (void)out_size; (void)ws_size;
  const float* tokens     = (const float*)d_in[0];
  const float* seq_tokens = (const float*)d_in[1];
  const float* modulation = (const float*)d_in[2];
  const float* attn_norm_w= (const float*)d_in[3];
  const float* qkv_w      = (const float*)d_in[4];
  const float* qkv_b      = (const float*)d_in[5];
  const float* out_w      = (const float*)d_in[6];
  const float* out_b      = (const float*)d_in[7];
  const float* film_w     = (const float*)d_in[8];
  const float* film_b     = (const float*)d_in[9];
  const float* sq_norm_w  = (const float*)d_in[10];
  const float* s_norm_w   = (const float*)d_in[11];
  const float* mha_in_w   = (const float*)d_in[12];
  const float* mha_in_b   = (const float*)d_in[13];
  const float* mha_out_w  = (const float*)d_in[14];
  const float* mha_out_b  = (const float*)d_in[15];
  const float* gate_w     = (const float*)d_in[16];
  const float* gate_b     = (const float*)d_in[17];
  const float* ffn_norm_w = (const float*)d_in[18];
  const float* gu_w       = (const float*)d_in[19];
  const float* gu_b       = (const float*)d_in[20];
  const float* down_w     = (const float*)d_in[21];
  const float* down_b     = (const float*)d_in[22];
  const unsigned char* padding_mask = (const unsigned char*)d_in[23];
  const unsigned char* seq_mask     = (const unsigned char*)d_in[24];
  float* out = (float*)d_out;

  char* wsb = (char*)d_ws;
  size_t off = 0;
  auto alloc = [&](size_t bytes) -> void* {
    off = (off + 255) & ~(size_t)255;
    void* p = wsb + off;
    off += bytes;
    return p;
  };
  const size_t MD = (size_t)M_ * D_;

  // bf16 weights
  bf* w_qkv  = (bf*)alloc((size_t)3 * D_ * D_ * 2);
  bf* w_out  = (bf*)alloc((size_t)D_ * D_ * 2);
  bf* w_mha  = (bf*)alloc((size_t)3 * D_ * D_ * 2);
  bf* w_mho  = (bf*)alloc((size_t)D_ * D_ * 2);
  bf* w_gate = (bf*)alloc((size_t)D_ * 2 * D_ * 2);
  bf* w_gu   = (bf*)alloc((size_t)2 * HID_ * D_ * 2);
  bf* w_down = (bf*)alloc((size_t)D_ * HID_ * 2);
  // bf16 activations (lifetime-reused)
  bf* BQ1 = (bf*)alloc(MD * 2);              // x_bf -> query_bf -> h_bf
  bf* BQ2 = (bf*)alloc(MD * 2);              // q_bf -> src_bf -> att2_bf
  bf* BQ3 = (bf*)alloc(MD * 2);              // k_bf -> cqh -> upd_bf
  bf* BQ4 = (bf*)alloc(MD * 2);              // vT_bf -> ckh
  bf* BQ5 = (bf*)alloc(MD * 2);              // attn_bf -> cvT
  bf* BQ6 = (bf*)alloc(MD * 2);              // tok2_bf
  bf* FF  = (bf*)alloc((size_t)M_ * HID_ * 2); // ff_bf
  // f32 buffers
  float* FP1  = (float*)alloc((size_t)M_ * HID_ * 4); // qkv_f / cq+ckv / val_f
  float* tok2 = (float*)alloc(MD * 4);
  float* updf = (float*)alloc(MD * 4);
  float* G1   = (float*)alloc(MD * 4);
  float* tok3 = (float*)alloc(MD * 4);
  float* ss     = (float*)alloc((size_t)B_ * 2 * D_ * 4);
  float* vself  = (float*)alloc((size_t)B_ * S_LEN * 4);
  float* vseq   = (float*)alloc((size_t)B_ * L_LEN * 4);
  float* vrows  = (float*)alloc(B_ * 4);

  auto conv = [&](const float* s, bf* d, size_t n) {
    f32_to_bf16_kernel<<<(unsigned)((n + 255) / 256), 256, 0, stream>>>(s, d, n);
  };
  auto gemm = [&](const bf* A, int lda, const bf* W, int ldw, const float* bias,
                  float* Cf, bf* Cb, int ldc, const float* res, const float* aux,
                  const float* aux2, const float* rs, int rpb, int Mm, int Nn, int Kk,
                  int epi) {
    dim3 grid(Nn / 64, Mm / 128);
    gemm_bf16_kernel<<<grid, 128, 0, stream>>>(A, lda, W, ldw, bias, Cf, Cb, ldc, res,
                                               aux, aux2, rs, rpb, Kk, epi);
  };

  // 0) weights -> bf16
  conv(qkv_w, w_qkv, (size_t)3 * D_ * D_);
  conv(out_w, w_out, (size_t)D_ * D_);
  conv(mha_in_w, w_mha, (size_t)3 * D_ * D_);
  conv(mha_out_w, w_mho, (size_t)D_ * D_);
  conv(gate_w, w_gate, (size_t)D_ * 2 * D_);
  conv(gu_w, w_gu, (size_t)2 * HID_ * D_);
  conv(down_w, w_down, (size_t)D_ * HID_);

  // 1) FiLM + masks
  film_kernel<<<dim3(2 * D_, B_), 256, 0, stream>>>(modulation, film_w, film_b, ss);
  mask_kernel<<<B_, 256, 0, stream>>>(padding_mask, vself, nullptr, S_LEN);
  mask_kernel<<<B_, 256, 0, stream>>>(seq_mask, vseq, vrows, L_LEN);

  // 2) self-attention branch
  rmsnorm_kernel<<<M_, 256, 0, stream>>>(tokens, attn_norm_w, ss, BQ1, S_LEN);
  gemm(BQ1, D_, w_qkv, D_, qkv_b, FP1, nullptr, 3 * D_, nullptr, nullptr, nullptr,
       nullptr, 0, M_, 3 * D_, D_, 0);
  {
    size_t total = (size_t)B_ * S_LEN * H_ * (HD_ / 2);
    rope_split_kernel<<<(unsigned)(total / 256), 256, 0, stream>>>(FP1, BQ2, BQ3, BQ4,
                                                                   total);
  }
  flash_attn_kernel<<<dim3(S_LEN / 64, H_, B_), 128, 0, stream>>>(BQ2, BQ3, BQ4, vself,
                                                                  BQ5, S_LEN, S_LEN);
  // tokens2 = tokens + attn @ out_w^T + out_b   (+ bf16 copy)
  gemm(BQ5, D_, w_out, D_, out_b, tok2, BQ6, D_, tokens, nullptr, nullptr, nullptr, 0,
       M_, D_, D_, 2);

  // 3) cross-attention branch
  rmsnorm_kernel<<<M_, 256, 0, stream>>>(tok2, sq_norm_w, nullptr, BQ1, S_LEN);
  rmsnorm_kernel<<<M_, 256, 0, stream>>>(seq_tokens, s_norm_w, nullptr, BQ2, L_LEN);
  float* cq = FP1;
  float* ckv = FP1 + MD;
  gemm(BQ1, D_, w_mha, D_, mha_in_b, cq, nullptr, D_, nullptr, nullptr, nullptr,
       nullptr, 0, M_, D_, D_, 0);
  gemm(BQ2, D_, w_mha + (size_t)D_ * D_, D_, mha_in_b + D_, ckv, nullptr, 2 * D_,
       nullptr, nullptr, nullptr, nullptr, 0, M_, 2 * D_, D_, 0);
  {
    size_t total = (size_t)B_ * L_LEN * H_ * HD_;
    cross_split_kernel<<<(unsigned)(total / 256), 256, 0, stream>>>(cq, ckv, BQ3, BQ4,
                                                                    BQ5, total);
  }
  flash_attn_kernel<<<dim3(S_LEN / 64, H_, B_), 128, 0, stream>>>(BQ3, BQ4, BQ5, vseq,
                                                                  BQ2, S_LEN, L_LEN);
  // update = (attended @ mha_out^T + b) * valid_rows[b]
  gemm(BQ2, D_, w_mho, D_, mha_out_b, updf, BQ3, D_, nullptr, nullptr, nullptr, vrows,
       S_LEN, M_, D_, D_, 3);

  // 4) gated residual: two-pass K-split GEMM over concat([tokens2, update])
  gemm(BQ6, D_, w_gate, 2 * D_, nullptr, G1, nullptr, D_, nullptr, nullptr, nullptr,
       nullptr, 0, M_, D_, D_, 0);
  gemm(BQ3, D_, w_gate + D_, 2 * D_, gate_b, tok3, nullptr, D_, tok2, G1, updf,
       nullptr, 0, M_, D_, D_, 4);

  // 5) FFN
  rmsnorm_kernel<<<M_, 256, 0, stream>>>(tok3, ffn_norm_w, nullptr, BQ1, S_LEN);
  float* valf = FP1;
  gemm(BQ1, D_, w_gu + (size_t)HID_ * D_, D_, gu_b + HID_, valf, nullptr, HID_,
       nullptr, nullptr, nullptr, nullptr, 0, M_, HID_, D_, 0);
  gemm(BQ1, D_, w_gu, D_, gu_b, nullptr, FF, HID_, nullptr, valf, nullptr, nullptr, 0,
       M_, HID_, D_, 5);
  // out = tokens3 + ff @ down_w^T + down_b
  gemm(FF, HID_, w_down, HID_, down_b, out, nullptr, D_, tok3, nullptr, nullptr,
       nullptr, 0, M_, D_, HID_, 2);
}
